// FEM_35072702939287
// MI455X (gfx1250) — compile-verified, matrix operand order checked by reading, not run on hardware
//
#include <hip/hip_runtime.h>
#include <cstdint>
#include <cstddef>

typedef __attribute__((ext_vector_type(16))) _Float16 v16h;
typedef __attribute__((ext_vector_type(8)))  _Float16 v8h;
typedef __attribute__((ext_vector_type(8)))  float    v8f;

#define NN   16
#define CCH  64
#define ICH  32
#define TVN  1600
#define NT16 (TVN / 16)     // 100 16-token tiles per batch
#define NT32 (TVN / 32)     // 50 32-token K-tiles per batch
#define SROWH 1608          // padded LDS row stride in halves (3216 B, 16B-mult)
#define EPSB 1e-5f

// Fragment-layout helpers (CDNA5 ISA 7.12.2, wave32, f16 16x16x32):
//  A 16x32: lane = M%16 + 16*h, h-half holds K in {kb..kb+7} u {kb+16..kb+23},
//           kb = h*8; element j<8 -> K=kb+j, j>=8 -> K=kb+16+(j-8).
//  B 32x16: identical with N in place of M.
//  C/D f32: N = lane&15, VGPR r -> M = r + (lane>>4)*8.
// For a channel/K index i in [0,32): h(i) = (i>>3)&1, j(i) = (i&7) + 8*(i>>4).

// ---------------------------------------------------------------------------
// Kernel 1: fused 1x1 projections -> PRE-SWIZZLED f16 WMMA operand buffers.
//   ksw: A-fragments for S = K*Q      [n][tile16][lane][16]   (token=M)
//   qsw: B-fragments for S = K*Q      [n][tile16][lane][16]   (token=N, chan=K)
//   vsw: B-fragments for P = expS*V   [n][tile32][ictile][lane][16] (token=K)
// ---------------------------------------------------------------------------
__global__ void proj_kernel(const float* __restrict__ x,
                            const float* __restrict__ Wv, const float* __restrict__ bv,
                            const float* __restrict__ Wk, const float* __restrict__ bk,
                            const float* __restrict__ Wq, const float* __restrict__ bq,
                            _Float16* __restrict__ ksw, _Float16* __restrict__ qsw,
                            _Float16* __restrict__ vsw) {
  int idx = blockIdx.x * blockDim.x + threadIdx.x;   // [0, NN*TVN)
  int n = idx / TVN, tv = idx % TVN;
  float xc[CCH];
  const float* xb = x + (size_t)n * CCH * TVN + tv;
  #pragma unroll
  for (int c = 0; c < CCH; ++c) xc[c] = xb[(size_t)c * TVN];

  const int tt = tv >> 4, pos = tv & 15;          // 16-token tile / position
  const int kt = tv >> 5, kpos = tv & 31;         // 32-token K-tile / position
  const int hV = (kpos >> 3) & 1;
  const int jV = (kpos & 7) + 8 * ((kpos >> 4) & 1);
  const size_t kqbase = ((size_t)n * NT16 + tt) * 512;          // 32 lanes * 16
  const size_t vbase  = ((size_t)n * NT32 + kt) * 1024;         // 2 ictiles * 512

  for (int i = 0; i < ICH; ++i) {
    float sv = bv[i], sk = bk[i], sq = bq[i];
    const float* wv = Wv + i * CCH;
    const float* wk = Wk + i * CCH;
    const float* wq = Wq + i * CCH;
    #pragma unroll
    for (int c = 0; c < CCH; ++c) {
      sv += wv[c] * xc[c];
      sk += wk[c] * xc[c];
      sq += wq[c] * xc[c];
    }
    const int h = (i >> 3) & 1;
    const int j = (i & 7) + 8 * (i >> 4);
    const int lane = pos + 16 * h;
    ksw[kqbase + lane * 16 + j] = (_Float16)sk;
    qsw[kqbase + lane * 16 + j] = (_Float16)sq;
    const int laneV = (i & 15) + 16 * hV;
    vsw[vbase + (size_t)(i >> 4) * 512 + laneV * 16 + jV] = (_Float16)sv;
  }
}

// ---------------------------------------------------------------------------
// Kernel 2: attention for one (batch, 16-row tile). 4 wave32 / block.
//  Phase 1: S = K_tile * Q (v_wmma_f32_16x16x32_f16, fragments = 32B/lane loads,
//           branchless global_prefetch of the next tile)
//  Phase 2: row softmax in LDS, both passes vectorized over 16B v8h chunks
//  Phase 3: P = expS * V, A-fragments via 2x ds_load_b128; P written straight
//           into B-fragment layout (psw) for the Wt WMMA stage.
// ---------------------------------------------------------------------------
__global__ void __launch_bounds__(128)
attn_kernel(const _Float16* __restrict__ ksw, const _Float16* __restrict__ qsw,
            const _Float16* __restrict__ vsw, _Float16* __restrict__ psw) {
  extern __shared__ char smem_raw[];
  _Float16* S = (_Float16*)smem_raw;                               // [16][SROWH]
  float* red = (float*)(smem_raw + 16 * SROWH * sizeof(_Float16)); // [16][8]
  float* rowstat = red + 128;                                      // max[16], sum[16]

  const int tid  = threadIdx.x;
  const int wave = tid >> 5, lane = tid & 31;
  const int m  = lane & 15;
  const int hh = lane >> 4;
  const int kb = hh * 8;
  const int n  = blockIdx.x / NT16;
  const int mt = blockIdx.x % NT16;

  const v16h* kswv = (const v16h*)ksw;
  const v16h* qswv = (const v16h*)qsw;
  const v16h* vswv = (const v16h*)vsw;

  // A fragment: K-tile rows (one 32B vector load per lane)
  const v16h afrag = kswv[((size_t)n * NT16 + mt) * 32 + lane];

  // ---- Phase 1: scores S[16][1600] ----
  for (int ct = wave; ct < NT16; ct += 4) {
    const size_t bidx = ((size_t)n * NT16 + ct) * 32 + lane;
    // branchless prefetch of the next tile this wave will consume
    const int ctn = (ct + 4 < NT16) ? (ct + 4) : ct;
    __builtin_prefetch(qswv + ((size_t)n * NT16 + ctn) * 32 + lane, 0, 1);
    const v16h bfrag = qswv[bidx];
    v8f c = {};
    c = __builtin_amdgcn_wmma_f32_16x16x32_f16(false, afrag, false, bfrag,
                                               (short)0, c, false, false);
    #pragma unroll
    for (int r = 0; r < 8; ++r) {
      int M = r + hh * 8;
      S[M * SROWH + ct * 16 + m] = (_Float16)c[r];
    }
  }
  __syncthreads();

  // ---- Phase 2: softmax, vectorized over 16B chunks ----
  const int r    = tid & 15;
  const int part = tid >> 4;
  const int sbeg = part * (TVN / 8);               // 200 halves = 25 x v8h
  v8h* rowv = (v8h*)(S + r * SROWH + sbeg);        // 16B-aligned (3216, 400)

  float mx = -3.4e38f;
  for (int it = 0; it < 25; ++it) {
    v8h vv = rowv[it];
    #pragma unroll
    for (int e = 0; e < 8; ++e) mx = fmaxf(mx, (float)vv[e]);
  }
  red[r * 8 + part] = mx;
  __syncthreads();
  if (part == 0) {
    float v = red[r * 8];
    #pragma unroll
    for (int j = 1; j < 8; ++j) v = fmaxf(v, red[r * 8 + j]);
    rowstat[r] = v;
  }
  __syncthreads();
  const float rm = rowstat[r];
  float sum = 0.f;
  for (int it = 0; it < 25; ++it) {
    v8h vv = rowv[it];
    v8h ee;
    #pragma unroll
    for (int e = 0; e < 8; ++e) {
      float e32 = __expf((float)vv[e] - rm);
      ee[e] = (_Float16)e32;
      sum += e32;
    }
    rowv[it] = ee;
  }
  red[r * 8 + part] = sum;
  __syncthreads();
  if (part == 0) {
    float v = 0.f;
    #pragma unroll
    for (int j = 0; j < 8; ++j) v += red[r * 8 + j];
    rowstat[16 + r] = v;
  }
  __syncthreads();

  // ---- Phase 3: P = expS * V  (waves 0/1 take the two 16-channel halves) ----
  if (wave < 2) {
    v8f acc = {};
    for (int kt = 0; kt < NT32; ++kt) {
      const int k0 = kt * 32;
      // A fragment from LDS: two aligned 16B loads, combined
      v8h lo = *(const v8h*)(S + m * SROWH + k0 + kb);
      v8h hi = *(const v8h*)(S + m * SROWH + k0 + kb + 16);
      v16h a2 = __builtin_shufflevector(lo, hi, 0, 1, 2, 3, 4, 5, 6, 7,
                                        8, 9, 10, 11, 12, 13, 14, 15);
      // B fragment: one 32B vector load per lane
      const v16h b2 = vswv[(((size_t)n * NT32 + kt) * 2 + wave) * 32 + lane];
      acc = __builtin_amdgcn_wmma_f32_16x16x32_f16(false, a2, false, b2,
                                                   (short)0, acc, false, false);
    }
    // Write P into B-fragment layout for the Wt stage.
    // This lane holds channel i = wave*16 + m for rows M = rr + hh*8.
    const int hB = (m >> 3) & 1;               // == (i>>3)&1 for both waves
    const int jB = (m & 7) + 8 * wave;         // == (i&7) + 8*(i>>4)
    _Float16* pb = psw + ((size_t)n * NT16 + mt) * 512;
    #pragma unroll
    for (int rr = 0; rr < 8; ++rr) {
      int M = rr + hh * 8;                     // token position in tile
      int laneB = M + 16 * hB;
      pb[laneB * 16 + jB] = (_Float16)(acc[rr] / rowstat[16 + M]);
    }
  }
}

// ---------------------------------------------------------------------------
// Kernel 3: Trans_s conv on WMMA:  y[n][o][t] = Wt[o,:] . P[n,:,t] + bt[o]
// 4 waves = 4 M-tiles (64 output channels); each block does 4 token tiles.
// ---------------------------------------------------------------------------
__global__ void __launch_bounds__(128)
trans_kernel(const _Float16* __restrict__ psw, const float* __restrict__ Wt,
             const float* __restrict__ bt, float* __restrict__ y) {
  const int tid  = threadIdx.x;
  const int wave = tid >> 5, lane = tid & 31;
  const int m  = lane & 15;
  const int hh = lane >> 4;
  const int n   = blockIdx.x / 25;
  const int ct0 = (blockIdx.x % 25) * 4;

  // A fragment: Wt rows o = wave*16 + m (K = ic channel), converted to f16
  const int o = wave * 16 + m;
  const float* wrow = Wt + o * ICH;
  v16h afrag;
  #pragma unroll
  for (int j = 0; j < 8; ++j) {
    afrag[j]     = (_Float16)wrow[hh * 8 + j];
    afrag[8 + j] = (_Float16)wrow[hh * 8 + 16 + j];
  }

  const v16h* pswv = (const v16h*)psw;
  #pragma unroll
  for (int c4 = 0; c4 < 4; ++c4) {
    const int ct = ct0 + c4;
    const v16h bfrag = pswv[((size_t)n * NT16 + ct) * 32 + lane];
    v8f c = {};
    c = __builtin_amdgcn_wmma_f32_16x16x32_f16(false, afrag, false, bfrag,
                                               (short)0, c, false, false);
    #pragma unroll
    for (int rr = 0; rr < 8; ++rr) {
      int oo = wave * 16 + rr + hh * 8;        // output channel (C/D row)
      int t  = ct * 16 + m;                    // token (C/D col)
      y[((size_t)n * CCH + oo) * TVN + t] = c[rr] + bt[oo];
    }
  }
}

// ---------------------------------------------------------------------------
// Kernel 4: per-channel batch mean/var over (N, T, V). One block per channel.
// ---------------------------------------------------------------------------
__global__ void stats_kernel(const float* __restrict__ y, float* __restrict__ stats) {
  __shared__ float ssum[256], ssq[256];
  const int c = blockIdx.x, tid = threadIdx.x;
  float s = 0.f, q = 0.f;
  for (int i = tid; i < NN * TVN; i += 256) {
    int n = i / TVN, tv = i % TVN;
    float v = y[((size_t)n * CCH + c) * TVN + tv];
    s += v; q += v * v;
  }
  ssum[tid] = s; ssq[tid] = q;
  __syncthreads();
  for (int off = 128; off > 0; off >>= 1) {
    if (tid < off) { ssum[tid] += ssum[tid + off]; ssq[tid] += ssq[tid + off]; }
    __syncthreads();
  }
  if (tid == 0) {
    float inv = 1.0f / (NN * TVN);
    float mean = ssum[0] * inv;
    stats[c]       = mean;
    stats[CCH + c] = ssq[0] * inv - mean * mean;
  }
}

// ---------------------------------------------------------------------------
// Kernel 5: ChannelGate — avg pool over (T,V) + MLP 64->4->64 + sigmoid.
// ---------------------------------------------------------------------------
__global__ void gate_kernel(const float* __restrict__ x,
                            const float* __restrict__ W1, const float* __restrict__ b1,
                            const float* __restrict__ W2, const float* __restrict__ b2,
                            float* __restrict__ gate) {
  __shared__ float part[256];
  __shared__ float avg[CCH];
  __shared__ float hid[4];
  const int n = blockIdx.x, tid = threadIdx.x;
  const int c = tid >> 2, q = tid & 3;
  float s = 0.f;
  const float* xb = x + ((size_t)n * CCH + c) * TVN;
  for (int i = q * (TVN / 4); i < (q + 1) * (TVN / 4); ++i) s += xb[i];
  part[tid] = s;
  __syncthreads();
  if (q == 0)
    avg[c] = (part[tid] + part[tid + 1] + part[tid + 2] + part[tid + 3]) * (1.0f / TVN);
  __syncthreads();
  if (tid < 4) {
    float h = b1[tid];
    for (int cc = 0; cc < CCH; ++cc) h += W1[tid * CCH + cc] * avg[cc];
    hid[tid] = fmaxf(h, 0.f);
  }
  __syncthreads();
  if (tid < CCH) {
    float g = b2[tid];
    #pragma unroll
    for (int j = 0; j < 4; ++j) g += W2[tid * 4 + j] * hid[j];
    gate[n * CCH + tid] = 1.f / (1.f + __expf(-g));
  }
}

// ---------------------------------------------------------------------------
// Kernel 6: out = gate * BN(y) + x   (fused epilogue)
// ---------------------------------------------------------------------------
__global__ void final_kernel(const float* __restrict__ x, const float* __restrict__ y,
                             const float* __restrict__ stats, const float* __restrict__ gate,
                             const float* __restrict__ gamma, const float* __restrict__ beta,
                             float* __restrict__ out) {
  int idx = blockIdx.x * blockDim.x + threadIdx.x;
  int c = (idx / TVN) % CCH;
  int n = idx / (TVN * CCH);
  float mean = stats[c], var = stats[CCH + c];
  float v = (y[idx] - mean) * rsqrtf(var + EPSB) * gamma[c] + beta[c];
  out[idx] = gate[n * CCH + c] * v + x[idx];
}

// ---------------------------------------------------------------------------
extern "C" void kernel_launch(void* const* d_in, const int* in_sizes, int n_in,
                              void* d_out, int out_size, void* d_ws, size_t ws_size,
                              hipStream_t stream) {
  const float* x     = (const float*)d_in[0];
  const float* Wv    = (const float*)d_in[1];
  const float* bv    = (const float*)d_in[2];
  const float* Wk    = (const float*)d_in[3];
  const float* bk    = (const float*)d_in[4];
  const float* Wq    = (const float*)d_in[5];
  const float* bq    = (const float*)d_in[6];
  const float* Wt    = (const float*)d_in[7];
  const float* bt    = (const float*)d_in[8];
  const float* gamma = (const float*)d_in[9];
  const float* beta  = (const float*)d_in[10];
  const float* W1    = (const float*)d_in[11];
  const float* b1    = (const float*)d_in[12];
  const float* W2    = (const float*)d_in[13];
  const float* b2    = (const float*)d_in[14];
  float* out = (float*)d_out;

  // Workspace layout (bytes); each swizzled operand buffer = N*TV*IC halves
  char* ws = (char*)d_ws;
  const size_t frag_bytes = (size_t)NN * TVN * ICH * sizeof(_Float16); // 1,638,400
  _Float16* ksw = (_Float16*)(ws);
  _Float16* qsw = (_Float16*)(ws + frag_bytes);
  _Float16* vsw = (_Float16*)(ws + 2 * frag_bytes);
  _Float16* psw = (_Float16*)(ws + 3 * frag_bytes);
  float*    y   = (float*)(ws + 4 * frag_bytes);                   // N*C*TV f32
  float*  stats = (float*)(ws + 4 * frag_bytes + (size_t)NN * CCH * TVN * 4);
  float*   gate = stats + 2 * CCH;

  // 1) projections -> pre-swizzled f16 WMMA fragments
  proj_kernel<<<(NN * TVN) / 256, 256, 0, stream>>>(x, Wv, bv, Wk, bk, Wq, bq,
                                                    ksw, qsw, vsw);
  // 2) attention (S -> softmax -> P), one block per (batch, 16-row tile)
  size_t smem = (size_t)16 * SROWH * sizeof(_Float16) + (128 + 32) * sizeof(float);
  attn_kernel<<<NN * NT16, 128, smem, stream>>>(ksw, qsw, vsw, psw);
  // 3) Trans_s conv on WMMA
  trans_kernel<<<NN * 25, 128, 0, stream>>>(psw, Wt, bt, y);
  // 4) batch-norm statistics
  stats_kernel<<<CCH, 256, 0, stream>>>(y, stats);
  // 5) channel gate
  gate_kernel<<<NN, 256, 0, stream>>>(x, W1, b1, W2, b2, gate);
  // 6) fused BN + gate + residual
  final_kernel<<<(NN * CCH * TVN) / 256, 256, 0, stream>>>(x, y, stats, gate,
                                                           gamma, beta, out);
}
